// QuantumTransformerDenoiser_12524124635737
// MI455X (gfx1250) — matrix operand order repaired and checked by reading, not compiled, not run on hardware
//
#include <hip/hip_runtime.h>
#include <hip/hip_bf16.h>

// ---------------------------------------------------------------------------
// QuantumTransformerDenoiser for MI455X (gfx1250, wave32)
//  - WMMA bf16 16x16x32 for all GEMMs; weights pre-converted to bf16 so the
//    inner loop is pure b128 loads + v_wmma (no cvt); 4 m-tiles per wave for
//    4x B-fragment reuse per WMMA operand load
//  - one-wave-per-token register-resident 6-qubit statevector (shuffle gates)
//  - streaming online-softmax MHSA and RBF kernel attention (+ prefetch)
// ---------------------------------------------------------------------------

typedef __bf16 bf16_t;
typedef __attribute__((ext_vector_type(16))) __bf16 bf16x16;
typedef __attribute__((ext_vector_type(8)))  __bf16 bf16x8;
typedef __attribute__((ext_vector_type(8)))  float  f32x8;

#define BB 16
#define LL 1024
#define NN (BB * LL)

// ---------------- wave32 reductions ----------------
__device__ __forceinline__ float wred_sum(float v) {
  v += __shfl_xor(v, 16, 32);
  v += __shfl_xor(v, 8, 32);
  v += __shfl_xor(v, 4, 32);
  v += __shfl_xor(v, 2, 32);
  v += __shfl_xor(v, 1, 32);
  return v;
}
__device__ __forceinline__ float wred_max(float v) {
  v = fmaxf(v, __shfl_xor(v, 16, 32));
  v = fmaxf(v, __shfl_xor(v, 8, 32));
  v = fmaxf(v, __shfl_xor(v, 4, 32));
  v = fmaxf(v, __shfl_xor(v, 2, 32));
  v = fmaxf(v, __shfl_xor(v, 1, 32));
  return v;
}

// ---------------- fp32 -> bf16 weight conversion (run once per weight) -------
__global__ void k_f2bf(const float* __restrict__ in, bf16_t* __restrict__ out, int n) {
  int i = blockIdx.x * blockDim.x + threadIdx.x;
  if (i < n) out[i] = (bf16_t)in[i];
}

// ---------------- unfold + amp-normalize + LN1 (1 wave / token) ----------------
__global__ void k_unfold_ln1(const float* __restrict__ x,
                             const float* __restrict__ g, const float* __restrict__ be,
                             bf16_t* __restrict__ patch_bf, float* __restrict__ t64,
                             bf16_t* __restrict__ hin) {
  const int tok = blockIdx.x, lane = threadIdx.x;
  const int b = tok >> 10, li = tok & 1023, hp = li >> 5, wp = li & 31;
  float p[2], t[2];
#pragma unroll
  for (int s = 0; s < 2; ++s) {
    int e = 2 * lane + s;
    int c = e >> 4, py = (e >> 2) & 3, px = e & 3;
    p[s] = x[(((size_t)(b * 4 + c) * 128) + (size_t)(hp * 4 + py)) * 128 + wp * 4 + px];
  }
  float nrm = sqrtf(wred_sum(p[0] * p[0] + p[1] * p[1]));
  t[0] = p[0] / (nrm + 1e-9f);
  t[1] = p[1] / (nrm + 1e-9f);
  float mean = wred_sum(t[0] + t[1]) * (1.0f / 64.0f);
  float d0 = t[0] - mean, d1 = t[1] - mean;
  float var = wred_sum(d0 * d0 + d1 * d1) * (1.0f / 64.0f);
  float inv = rsqrtf(var + 1e-5f);
#pragma unroll
  for (int s = 0; s < 2; ++s) {
    int e = 2 * lane + s;
    size_t o = (size_t)tok * 64 + e;
    patch_bf[o] = (bf16_t)p[s];
    t64[o] = t[s];
    float dv = s ? d1 : d0;
    hin[o] = (bf16_t)(dv * inv * g[e] + be[e]);
  }
}

// ---------------- generic LN -> bf16 (1 wave / token) ----------------
__global__ void k_ln(const float* __restrict__ in, const float* __restrict__ g,
                     const float* __restrict__ be, bf16_t* __restrict__ out) {
  const int tok = blockIdx.x, lane = threadIdx.x;
  size_t o = (size_t)tok * 64 + 2 * lane;
  float x0 = in[o], x1 = in[o + 1];
  float mean = wred_sum(x0 + x1) * (1.0f / 64.0f);
  float d0 = x0 - mean, d1 = x1 - mean;
  float var = wred_sum(d0 * d0 + d1 * d1) * (1.0f / 64.0f);
  float inv = rsqrtf(var + 1e-5f);
  out[o] = (bf16_t)(d0 * inv * g[2 * lane] + be[2 * lane]);
  out[o + 1] = (bf16_t)(d1 * inv * g[2 * lane + 1] + be[2 * lane + 1]);
}

// ---------------- WMMA bf16 GEMM: out[N,Dout] = A[N,K] @ Wb[Dout,K]^T -------------
// One wave computes a 64x16 strip (4 m-tiles) so each B fragment feeds 4 WMMAs.
// A and Wb are bf16 row-major; inner loop is b128 loads + v_wmma only.
__global__ void k_gemm_wmma(const bf16_t* __restrict__ A, int K,
                            const bf16_t* __restrict__ Wb,
                            const float* __restrict__ bias,
                            const float* __restrict__ resid, int res_stride,
                            float* __restrict__ outf, int out_stride,
                            bf16_t* __restrict__ outb, int outb_stride,
                            int act) {
  const int lane = threadIdx.x;
  const int m0 = blockIdx.x * 64;        // 4 m-tiles of 16 rows
  const int n0 = blockIdx.y * 16;
  const int khalf = lane >> 4;
  const int l15 = lane & 15;

  f32x8 c0 = {}, c1 = {}, c2 = {}, c3 = {};
  for (int kk = 0; kk < K; kk += 32) {
    // B fragment: lane column n = l15; halfs e: K = (e&7) + 8*khalf + 16*(e>>3)
    const bf16_t* wrow = Wb + (size_t)(n0 + l15) * K + kk + 8 * khalf;
    bf16x8 blo = *(const bf16x8*)(wrow);
    bf16x8 bhi = *(const bf16x8*)(wrow + 16);
    bf16x16 bf;
#pragma unroll
    for (int e = 0; e < 8; ++e) { bf[e] = blo[e]; bf[8 + e] = bhi[e]; }

    // A fragments for 4 m-tiles: halfs e: K = (e&7) + 8*khalf + 16*(e>>3)
#pragma unroll
    for (int t = 0; t < 4; ++t) {
      const bf16_t* arow = A + (size_t)(m0 + t * 16 + l15) * K + kk;
      bf16x8 alo = *(const bf16x8*)(arow + 8 * khalf);
      bf16x8 ahi = *(const bf16x8*)(arow + 16 + 8 * khalf);
      bf16x16 a;
#pragma unroll
      for (int e = 0; e < 8; ++e) { a[e] = alo[e]; a[8 + e] = ahi[e]; }
      if (t == 0)
        c0 = __builtin_amdgcn_wmma_f32_16x16x32_bf16(false, a, false, bf, (short)0, c0,
                                                     false, false);
      else if (t == 1)
        c1 = __builtin_amdgcn_wmma_f32_16x16x32_bf16(false, a, false, bf, (short)0, c1,
                                                     false, false);
      else if (t == 2)
        c2 = __builtin_amdgcn_wmma_f32_16x16x32_bf16(false, a, false, bf, (short)0, c2,
                                                     false, false);
      else
        c3 = __builtin_amdgcn_wmma_f32_16x16x32_bf16(false, a, false, bf, (short)0, c3,
                                                     false, false);
    }
  }

  const int n = n0 + l15;
  const float bv = bias ? bias[n] : 0.0f;
#pragma unroll
  for (int t = 0; t < 4; ++t) {
    f32x8 c = (t == 0) ? c0 : (t == 1) ? c1 : (t == 2) ? c2 : c3;
#pragma unroll
    for (int r = 0; r < 8; ++r) {
      int m = m0 + t * 16 + r + 8 * khalf;
      float v = c[r] + bv;
      if (act == 1) {  // tanh-approx GELU
        float xx = v;
        v = 0.5f * xx * (1.0f + tanhf(0.7978845608f * (xx + 0.044715f * xx * xx * xx)));
      }
      if (resid) v += resid[(size_t)m * res_stride + n];
      if (outf) outf[(size_t)m * out_stride + n] = v;
      if (outb) outb[(size_t)m * outb_stride + n] = (bf16_t)v;
    }
  }
}

// ---------------- classic MHSA, head dim 8, online softmax (1 wave / (b,h,i)) ---------
__global__ void k_classic_attn(const float* __restrict__ Q, const float* __restrict__ Kb,
                               const float* __restrict__ V, bf16_t* __restrict__ out) {
  const int blk = blockIdx.x, lane = threadIdx.x;
  const int i = blk & 1023, h = (blk >> 10) & 7, b = blk >> 13;
  const size_t rowq = ((size_t)(b * 1024 + i)) * 64 + h * 8;
  float q[8];
#pragma unroll
  for (int d = 0; d < 8; ++d) q[d] = Q[rowq + d];

  float m = -3.0e38f, l = 0.0f;
  float acc[8] = {0, 0, 0, 0, 0, 0, 0, 0};
  for (int j = lane; j < 1024; j += 32) {
    const size_t rk = ((size_t)(b * 1024 + j)) * 64 + h * 8;
    if (j + 32 < 1024) __builtin_prefetch(&Kb[rk + (size_t)32 * 64], 0, 1);
    float4 k0 = ((const float4*)(Kb + rk))[0];
    float4 k1 = ((const float4*)(Kb + rk))[1];
    float s = q[0] * k0.x + q[1] * k0.y + q[2] * k0.z + q[3] * k0.w +
              q[4] * k1.x + q[5] * k1.y + q[6] * k1.z + q[7] * k1.w;
    s *= 0.3535533906f;  // 1/sqrt(8)
    float mn = fmaxf(m, s);
    float f = __expf(m - mn), e = __expf(s - mn);
    l = l * f + e;
    float4 v0 = ((const float4*)(V + rk))[0];
    float4 v1 = ((const float4*)(V + rk))[1];
    acc[0] = acc[0] * f + e * v0.x; acc[1] = acc[1] * f + e * v0.y;
    acc[2] = acc[2] * f + e * v0.z; acc[3] = acc[3] * f + e * v0.w;
    acc[4] = acc[4] * f + e * v1.x; acc[5] = acc[5] * f + e * v1.y;
    acc[6] = acc[6] * f + e * v1.z; acc[7] = acc[7] * f + e * v1.w;
    m = mn;
  }
  float mg = wred_max(m);
  float f = __expf(m - mg);
  l = wred_sum(l * f);
#pragma unroll
  for (int d = 0; d < 8; ++d) acc[d] = wred_sum(acc[d] * f);
  if (lane == 0) {
#pragma unroll
    for (int d = 0; d < 8; ++d) out[rowq + d] = (bf16_t)(acc[d] / l);
  }
}

// ---------------- 6-qubit statevector machinery (2 amps / lane) ----------------
struct Amp { float r0, i0, r1, i1; };  // lane owns amplitudes 2*lane, 2*lane+1

__device__ __forceinline__ void gate_rx(Amp& a, float t, int q, int lane) {
  float s, c;
  __sincosf(0.5f * t, &s, &c);
  const int bpos = 5 - q;
  float pr0, pi0, pr1, pi1;
  if (bpos == 0) {
    pr0 = a.r1; pi0 = a.i1; pr1 = a.r0; pi1 = a.i0;
  } else {
    const int msk = 1 << (bpos - 1);
    pr0 = __shfl_xor(a.r0, msk, 32); pi0 = __shfl_xor(a.i0, msk, 32);
    pr1 = __shfl_xor(a.r1, msk, 32); pi1 = __shfl_xor(a.i1, msk, 32);
  }
  a.r0 = c * a.r0 + s * pi0; a.i0 = c * a.i0 - s * pr0;
  a.r1 = c * a.r1 + s * pi1; a.i1 = c * a.i1 - s * pr1;
}

__device__ __forceinline__ void gate_ry(Amp& a, float t, int q, int lane) {
  float s, c;
  __sincosf(0.5f * t, &s, &c);
  const int bpos = 5 - q;
  float pr0, pi0, pr1, pi1, sg0, sg1;
  if (bpos == 0) {
    pr0 = a.r1; pi0 = a.i1; pr1 = a.r0; pi1 = a.i0;
    sg0 = -s; sg1 = s;
  } else {
    const int msk = 1 << (bpos - 1);
    pr0 = __shfl_xor(a.r0, msk, 32); pi0 = __shfl_xor(a.i0, msk, 32);
    pr1 = __shfl_xor(a.r1, msk, 32); pi1 = __shfl_xor(a.i1, msk, 32);
    const float sg = ((lane >> (bpos - 1)) & 1) ? s : -s;
    sg0 = sg; sg1 = sg;
  }
  a.r0 = c * a.r0 + sg0 * pr0; a.i0 = c * a.i0 + sg0 * pi0;
  a.r1 = c * a.r1 + sg1 * pr1; a.i1 = c * a.i1 + sg1 * pi1;
}

__device__ __forceinline__ void gate_cnot(Amp& a, int i, int lane) {
  const int tb = 4 - i;
  const int cb = (lane >> (4 - i)) & 1;
  if (tb == 0) {
    if (cb) {
      float tr = a.r0; a.r0 = a.r1; a.r1 = tr;
      float ti = a.i0; a.i0 = a.i1; a.i1 = ti;
    }
  } else {
    const int msk = 1 << (tb - 1);
    float pr0 = __shfl_xor(a.r0, msk, 32), pi0 = __shfl_xor(a.i0, msk, 32);
    float pr1 = __shfl_xor(a.r1, msk, 32), pi1 = __shfl_xor(a.i1, msk, 32);
    a.r0 = cb ? pr0 : a.r0; a.i0 = cb ? pi0 : a.i0;
    a.r1 = cb ? pr1 : a.r1; a.i1 = cb ? pi1 : a.i1;
  }
}

__device__ void pqc(Amp& a, const float* __restrict__ w, int depth, int lane) {
  for (int l = 0; l < depth; ++l) {
    const float* wl = w + l * 18;
    for (int q = 0; q < 6; ++q) {
      gate_rx(a, wl[q * 3 + 0], q, lane);
      gate_ry(a, wl[q * 3 + 1], q, lane);
    }
    for (int q = 0; q < 5; ++q) gate_cnot(a, q, lane);
    for (int q = 0; q < 6; ++q) gate_ry(a, wl[q * 3 + 2], q, lane);
  }
}

__device__ __forceinline__ void measure_z(const Amp& a, int lane, int tok,
                                          float* __restrict__ z) {
  float p0 = a.r0 * a.r0 + a.i0 * a.i0;
  float p1 = a.r1 * a.r1 + a.i1 * a.i1;
#pragma unroll
  for (int w = 0; w < 6; ++w) {
    float contrib;
    if (w < 5) {
      float sg = 1.0f - 2.0f * (float)((lane >> (4 - w)) & 1);
      contrib = sg * (p0 + p1);
    } else {
      contrib = p0 - p1;
    }
    float zz = wred_sum(contrib);
    if (lane == 0) z[(size_t)tok * 6 + w] = zz;
  }
}

// ---------------- quantum attention front-end (1 wave / token) ----------------
__global__ void k_quantum(const float* __restrict__ h, const float* __restrict__ ln2_g,
                          const float* __restrict__ ln2_b,
                          const float* __restrict__ inp_scale,
                          const float* __restrict__ enc_w, const float* __restrict__ q_w,
                          const float* __restrict__ k_w, const float* __restrict__ v_w,
                          const float* __restrict__ mq_w, const float* __restrict__ mk_w,
                          const float* __restrict__ mv_w, float* __restrict__ zq,
                          float* __restrict__ zk, float* __restrict__ v64) {
  const int tok = blockIdx.x, lane = threadIdx.x;
  const size_t o = (size_t)tok * 64 + 2 * lane;
  float x0 = h[o], x1 = h[o + 1];
  float mean = wred_sum(x0 + x1) * (1.0f / 64.0f);
  float d0 = x0 - mean, d1 = x1 - mean;
  float var = wred_sum(d0 * d0 + d1 * d1) * (1.0f / 64.0f);
  float inv = rsqrtf(var + 1e-5f);
  float s0 = (d0 * inv * ln2_g[2 * lane] + ln2_b[2 * lane]) * inp_scale[2 * lane];
  float s1 = (d1 * inv * ln2_g[2 * lane + 1] + ln2_b[2 * lane + 1]) * inp_scale[2 * lane + 1];
  float nrm = sqrtf(wred_sum(s0 * s0 + s1 * s1));
  s0 /= (nrm + 1e-9f);
  s1 /= (nrm + 1e-9f);

  Amp enc;
  enc.r0 = s0; enc.i0 = 0.0f; enc.r1 = s1; enc.i1 = 0.0f;
  pqc(enc, enc_w, 2, lane);

  Amp aq = enc;
  pqc(aq, q_w, 2, lane);
  pqc(aq, mq_w, 1, lane);
  measure_z(aq, lane, tok, zq);

  Amp ak = enc;
  pqc(ak, k_w, 2, lane);
  pqc(ak, mk_w, 1, lane);
  measure_z(ak, lane, tok, zk);

  Amp av = enc;
  pqc(av, v_w, 2, lane);
  pqc(av, mv_w, 1, lane);
  {
    float p0 = av.r0 * av.r0 + av.i0 * av.i0;
    float p1 = av.r1 * av.r1 + av.i1 * av.i1;
    float S = wred_sum(p0 + p1);
    float invS = 1.0f / (S + 1e-9f);
    v64[o] = p0 * invS;
    v64[o + 1] = p1 * invS;
  }
}

// ---------------- z -> 4-dim projection + LN (256 thr, 1 token / thread) ----------------
__global__ void k_qkproj(const float* __restrict__ z, const float* __restrict__ pw,
                         const float* __restrict__ pb, const float* __restrict__ g,
                         const float* __restrict__ be, float* __restrict__ out) {
  const int tok = blockIdx.x * blockDim.x + threadIdx.x;
  if (tok >= NN) return;
  float y[4], mean = 0.0f;
#pragma unroll
  for (int oidx = 0; oidx < 4; ++oidx) {
    float acc = pb[oidx];
#pragma unroll
    for (int w = 0; w < 6; ++w) acc += z[(size_t)tok * 6 + w] * pw[oidx * 6 + w];
    y[oidx] = acc;
    mean += acc;
  }
  mean *= 0.25f;
  float var = 0.0f;
#pragma unroll
  for (int oidx = 0; oidx < 4; ++oidx) {
    float d = y[oidx] - mean;
    var += d * d;
  }
  var *= 0.25f;
  float inv = rsqrtf(var + 1e-5f);
#pragma unroll
  for (int oidx = 0; oidx < 4; ++oidx)
    out[(size_t)tok * 4 + oidx] = (y[oidx] - mean) * inv * g[oidx] + be[oidx];
}

// ---------------- RBF kernel attention, h += wts @ v64 (1 wave / (b,i)) ----------------
__global__ void k_rbf(const float* __restrict__ qv, const float* __restrict__ kv,
                      const float* __restrict__ v64, const float* __restrict__ raw_tau,
                      float* __restrict__ h) {
  const int blk = blockIdx.x, lane = threadIdx.x;
  const int b = blk >> 10;
  const float tau = log1pf(__expf(raw_tau[0]));
  const float invtau = 1.0f / tau;
  float q0 = qv[(size_t)blk * 4 + 0], q1 = qv[(size_t)blk * 4 + 1];
  float q2 = qv[(size_t)blk * 4 + 2], q3 = qv[(size_t)blk * 4 + 3];
  const float qq = q0 * q0 + q1 * q1 + q2 * q2 + q3 * q3;

  float S = 0.0f;
  float acc[64];
#pragma unroll
  for (int d = 0; d < 64; ++d) acc[d] = 0.0f;

  for (int j = lane; j < 1024; j += 32) {
    const size_t rk = (size_t)(b * 1024 + j);
    float4 kk = ((const float4*)(kv + rk * 4))[0];
    float d2 = qq + kk.x * kk.x + kk.y * kk.y + kk.z * kk.z + kk.w * kk.w -
               2.0f * (q0 * kk.x + q1 * kk.y + q2 * kk.z + q3 * kk.w);
    d2 = fmaxf(d2, 0.0f);
    float a = __expf(-d2 * invtau);
    S += a;
    const float4* vr = (const float4*)(v64 + rk * 64);
    if (j + 32 < 1024) __builtin_prefetch(v64 + (rk + 32) * 64, 0, 1);
#pragma unroll
    for (int t = 0; t < 16; ++t) {
      float4 vv = vr[t];
      acc[4 * t + 0] += a * vv.x;
      acc[4 * t + 1] += a * vv.y;
      acc[4 * t + 2] += a * vv.z;
      acc[4 * t + 3] += a * vv.w;
    }
  }
  S = wred_sum(S);
  const float invS = 1.0f / (S + 1e-9f);
#pragma unroll
  for (int d = 0; d < 64; ++d) acc[d] = wred_sum(acc[d]);
  if (lane == 0) {
    const size_t row = (size_t)blk * 64;
#pragma unroll
    for (int d = 0; d < 64; ++d) h[row + d] += acc[d] * invS;
  }
}

// ---------------------------------------------------------------------------
extern "C" void kernel_launch(void* const* d_in, const int* in_sizes, int n_in,
                              void* d_out, int out_size, void* d_ws, size_t ws_size,
                              hipStream_t stream) {
  (void)in_sizes; (void)n_in; (void)out_size; (void)ws_size;
  const float* x         = (const float*)d_in[0];
  const float* conv_w    = (const float*)d_in[1];
  const float* conv_b    = (const float*)d_in[2];
  const float* ln1_g     = (const float*)d_in[3];
  const float* ln1_b     = (const float*)d_in[4];
  const float* wq        = (const float*)d_in[5];
  const float* wk        = (const float*)d_in[6];
  const float* wv        = (const float*)d_in[7];
  const float* wo        = (const float*)d_in[8];
  const float* bo        = (const float*)d_in[9];
  const float* ln2_g     = (const float*)d_in[10];
  const float* ln2_b     = (const float*)d_in[11];
  const float* enc_w     = (const float*)d_in[12];
  const float* q_w       = (const float*)d_in[13];
  const float* k_w       = (const float*)d_in[14];
  const float* v_w       = (const float*)d_in[15];
  const float* mq_w      = (const float*)d_in[16];
  const float* mk_w      = (const float*)d_in[17];
  const float* mv_w      = (const float*)d_in[18];
  const float* qproj_w   = (const float*)d_in[19];
  const float* qproj_b   = (const float*)d_in[20];
  const float* kproj_w   = (const float*)d_in[21];
  const float* kproj_b   = (const float*)d_in[22];
  const float* qkln_g    = (const float*)d_in[23];
  const float* qkln_b    = (const float*)d_in[24];
  const float* raw_tau   = (const float*)d_in[25];
  const float* inp_scale = (const float*)d_in[26];
  const float* ln3_g     = (const float*)d_in[27];
  const float* ln3_b     = (const float*)d_in[28];
  const float* mlp_w1    = (const float*)d_in[29];
  const float* mlp_b1    = (const float*)d_in[30];
  const float* mlp_w2    = (const float*)d_in[31];
  const float* mlp_b2    = (const float*)d_in[32];

  float* out = (float*)d_out;  // [B, L, 448] fp32: cols 0..383 = t384, 384..447 = h

  // workspace carve-out (256B aligned chunks)
  char* p = (char*)d_ws;
  auto take = [&](size_t bytes) -> char* {
    char* r = p;
    p += (bytes + 255) & ~(size_t)255;
    return r;
  };
  bf16_t* patch_bf = (bf16_t*)take((size_t)NN * 64 * sizeof(bf16_t));
  float*  t64      = (float*) take((size_t)NN * 64 * sizeof(float));
  bf16_t* hin      = (bf16_t*)take((size_t)NN * 64 * sizeof(bf16_t));
  float*  qbuf     = (float*) take((size_t)NN * 64 * sizeof(float));
  float*  kbuf     = (float*) take((size_t)NN * 64 * sizeof(float));
  float*  vbuf     = (float*) take((size_t)NN * 64 * sizeof(float));
  bf16_t* attn     = (bf16_t*)take((size_t)NN * 64 * sizeof(bf16_t));
  float*  hbuf     = (float*) take((size_t)NN * 64 * sizeof(float));
  float*  zqb      = (float*) take((size_t)NN * 6 * sizeof(float));
  float*  zkb      = (float*) take((size_t)NN * 6 * sizeof(float));
  float*  v64b     = (float*) take((size_t)NN * 64 * sizeof(float));
  float*  qvb      = (float*) take((size_t)NN * 4 * sizeof(float));
  float*  kvb      = (float*) take((size_t)NN * 4 * sizeof(float));
  bf16_t* hm       = (bf16_t*)take((size_t)NN * 64 * sizeof(bf16_t));
  bf16_t* act      = (bf16_t*)take((size_t)NN * 256 * sizeof(bf16_t));
  // bf16 weight mirrors (converted once per launch; tiny)
  bf16_t* conv_wb  = (bf16_t*)take((size_t)384 * 64 * sizeof(bf16_t));
  bf16_t* wqb      = (bf16_t*)take((size_t)64 * 64 * sizeof(bf16_t));
  bf16_t* wkb      = (bf16_t*)take((size_t)64 * 64 * sizeof(bf16_t));
  bf16_t* wvb      = (bf16_t*)take((size_t)64 * 64 * sizeof(bf16_t));
  bf16_t* wob      = (bf16_t*)take((size_t)64 * 64 * sizeof(bf16_t));
  bf16_t* w1b      = (bf16_t*)take((size_t)256 * 64 * sizeof(bf16_t));
  bf16_t* w2b      = (bf16_t*)take((size_t)64 * 256 * sizeof(bf16_t));

  // 0) convert weights fp32 -> bf16 (once, negligible)
  k_f2bf<<<(384 * 64 + 255) / 256, 256, 0, stream>>>(conv_w, conv_wb, 384 * 64);
  k_f2bf<<<(64 * 64 + 255) / 256, 256, 0, stream>>>(wq, wqb, 64 * 64);
  k_f2bf<<<(64 * 64 + 255) / 256, 256, 0, stream>>>(wk, wkb, 64 * 64);
  k_f2bf<<<(64 * 64 + 255) / 256, 256, 0, stream>>>(wv, wvb, 64 * 64);
  k_f2bf<<<(64 * 64 + 255) / 256, 256, 0, stream>>>(wo, wob, 64 * 64);
  k_f2bf<<<(256 * 64 + 255) / 256, 256, 0, stream>>>(mlp_w1, w1b, 256 * 64);
  k_f2bf<<<(64 * 256 + 255) / 256, 256, 0, stream>>>(mlp_w2, w2b, 64 * 256);

  // 1) unfold patches, amplitude-normalize, LN1
  k_unfold_ln1<<<NN, 32, 0, stream>>>(x, ln1_g, ln1_b, patch_bf, t64, hin);

  // 2) t384 = patch @ conv_w^T + conv_b  -> out[:, 0:384]  (WMMA)
  k_gemm_wmma<<<dim3(NN / 64, 384 / 16), 32, 0, stream>>>(
      patch_bf, 64, conv_wb, conv_b, nullptr, 0, out, 448, nullptr, 0, 0);

  // 3) Q/K/V projections (WMMA)
  k_gemm_wmma<<<dim3(NN / 64, 4), 32, 0, stream>>>(hin, 64, wqb, nullptr, nullptr, 0,
                                                   qbuf, 64, nullptr, 0, 0);
  k_gemm_wmma<<<dim3(NN / 64, 4), 32, 0, stream>>>(hin, 64, wkb, nullptr, nullptr, 0,
                                                   kbuf, 64, nullptr, 0, 0);
  k_gemm_wmma<<<dim3(NN / 64, 4), 32, 0, stream>>>(hin, 64, wvb, nullptr, nullptr, 0,
                                                   vbuf, 64, nullptr, 0, 0);

  // 4) classic MHSA (online softmax, streaming)
  k_classic_attn<<<BB * 8 * LL, 32, 0, stream>>>(qbuf, kbuf, vbuf, attn);

  // 5) h = t64 + attn @ wo^T + bo  (WMMA + residual)
  k_gemm_wmma<<<dim3(NN / 64, 4), 32, 0, stream>>>(attn, 64, wob, bo, t64, 64, hbuf, 64,
                                                   nullptr, 0, 0);

  // 6) quantum circuit front-end: zq, zk, v64
  k_quantum<<<NN, 32, 0, stream>>>(hbuf, ln2_g, ln2_b, inp_scale, enc_w, q_w, k_w, v_w,
                                   mq_w, mk_w, mv_w, zqb, zkb, v64b);

  // 7) 6->4 projections + LN
  k_qkproj<<<NN / 256, 256, 0, stream>>>(zqb, qproj_w, qproj_b, qkln_g, qkln_b, qvb);
  k_qkproj<<<NN / 256, 256, 0, stream>>>(zkb, kproj_w, kproj_b, qkln_g, qkln_b, kvb);

  // 8) RBF kernel attention: h += wts @ v64
  k_rbf<<<NN, 32, 0, stream>>>(qvb, kvb, v64b, raw_tau, hbuf);

  // 9) MLP: hm = LN3(h); h += gelu(hm@w1^T+b1)@w2^T+b2 -> out[:, 384:448]
  k_ln<<<NN, 32, 0, stream>>>(hbuf, ln3_g, ln3_b, hm);
  k_gemm_wmma<<<dim3(NN / 64, 16), 32, 0, stream>>>(hm, 64, w1b, mlp_b1, nullptr, 0,
                                                    nullptr, 0, act, 256, 1);
  k_gemm_wmma<<<dim3(NN / 64, 4), 32, 0, stream>>>(act, 256, w2b, mlp_b2, hbuf, 64,
                                                   out + 384, 448, nullptr, 0, 0);
}
